// PositionCloser_17884243820576
// MI455X (gfx1250) — matrix operand order repaired and checked by reading, not compiled, not run on hardware
//
#include <hip/hip_runtime.h>
#include <math.h>

#define B0N 64
#define B1N 16
#define CHUNK 64
#define HID 256
#define TLEN 8192
#define NGATE 1024           // 4*HID gate rows
#define ROWS_PER_BLK 32
#define NBLK 32              // 32 * 32 = 1024 batch rows
#define NTH 512              // 16 waves of 32 -> 4 waves/SIMD, ~256 VGPR budget
#define HSTR 264             // bf16 elements per h row (256 + pad, 16B aligned)

typedef __attribute__((ext_vector_type(16))) __bf16 v16bf;
typedef __attribute__((ext_vector_type(8)))  float  v8f;
typedef __attribute__((ext_vector_type(4)))  int    i4;   // trivial 16B vector
typedef __attribute__((ext_vector_type(2)))  float  f2;

union Frag {
    v16bf v;
    i4    q[2];
};

__device__ __forceinline__ float sigm(float x) {
    // v_exp + v_rcp based; avoids the IEEE-division expansion
    return __builtin_amdgcn_rcpf(1.0f + __expf(-x));
}
__device__ __forceinline__ float tanh_(float x) { return 2.0f * sigm(2.0f * x) - 1.0f; }

__device__ __forceinline__ unsigned short f2bf16(float f) {
    __bf16 b = (__bf16)f;                     // native cvt, RNE
    return __builtin_bit_cast(unsigned short, b);
}

// ---- prep: n_chunks = (T - max(open_slices)) / CHUNK ------------------------
__global__ void prep_meta(const int* __restrict__ open_slices, int* __restrict__ nch_out) {
    __shared__ int smax[32];
    int tid = threadIdx.x;
    int v = open_slices[tid];                    // exactly 1024 entries
    #pragma unroll
    for (int d = 16; d > 0; d >>= 1) {
        int o = __shfl_xor(v, d);
        v = (o > v) ? o : v;
    }
    if ((tid & 31) == 0) smax[tid >> 5] = v;
    __syncthreads();
    if (tid < 32) {
        int m = smax[tid];
        #pragma unroll
        for (int d = 16; d > 0; d >>= 1) {
            int o = __shfl_xor(m, d);
            m = (o > m) ? o : m;
        }
        if (tid == 0) nch_out[0] = (TLEN - m) / CHUNK;
    }
}

// ---- prep: W_hh f32 -> bf16 (row-major [n][k], B-fragment friendly) ---------
__global__ void prep_whh(const float* __restrict__ whh, unsigned short* __restrict__ wb) {
    int base = blockIdx.x * 1024 + threadIdx.x * 4;
    #pragma unroll
    for (int i = 0; i < 4; ++i) wb[base + i] = f2bf16(whh[base + i]);
}

// ---- main persistent LSTM kernel: one block = 32 batch rows -----------------
__global__ void __launch_bounds__(NTH, 1) __attribute__((amdgpu_waves_per_eu(4, 4)))
lstm_main(const int* __restrict__ inds, const float* __restrict__ input,
          const float* __restrict__ p, const float* __restrict__ ls_probs,
          const float* __restrict__ open_probs, const int* __restrict__ open_slices,
          const float* __restrict__ open_hx, const float* __restrict__ W_ih,
          const float* __restrict__ b_ih, const float* __restrict__ b_hh,
          const float* __restrict__ W_out, const float* __restrict__ b_out,
          const unsigned short* __restrict__ whhb, const int* __restrict__ nch_ptr,
          float* __restrict__ loss_part) {
    __shared__ __align__(16) unsigned short hlds[ROWS_PER_BLK * HSTR];   // h (bf16)
    __shared__ __align__(16) float aalds[CHUNK * ROWS_PER_BLK * 2];      // (a0,a1) per (t,m)
    __shared__ float wterm[CHUNK * ROWS_PER_BLK];                        // open_prob * term
    __shared__ float scratch[8 * ROWS_PER_BLK];                          // dot partials per wj
    __shared__ float lsA[ROWS_PER_BLK], opA[ROWS_PER_BLK], oblA[ROWS_PER_BLK], oslA[ROWS_PER_BLK];

    const int tid  = threadIdx.x;
    const int lane = tid & 31;
    const int wid  = tid >> 5;       // 0..15
    const int lm   = lane & 15;
    const int lh   = lane >> 4;      // 0/1
    const int mt   = wid >> 3;       // M-tile 0..1
    const int wj   = wid & 7;        // N-tile family 0..7
    const int blk  = blockIdx.x;
    const int row0 = blk * ROWS_PER_BLK;
    const int nch  = nch_ptr[0];

    // per-lane constants: input-proj weights + bias for this lane's 8 gate columns
    float wih0[8], wih1[8], bsum[8];
    #pragma unroll
    for (int s = 0; s < 8; ++s) {
        int n = (wj + 8 * s) * 16 + lm;
        wih0[s] = W_ih[n * 2 + 0];
        wih1[s] = W_ih[n * 2 + 1];
        bsum[s] = b_ih[n] + b_hh[n];
    }
    const int j0 = wj * 16 + lm;     // hidden indices owned by this lane
    const int j1 = j0 + 128;
    const float wo0 = W_out[j0], wo1 = W_out[j1];

    // c state fully in registers: creg[jj][r] -> row 16*mt + r + 8*lh, hidden j0/j1
    float creg[2][8];
    #pragma unroll
    for (int r = 0; r < 8; ++r) {
        int grow = row0 + 16 * mt + r + 8 * lh;
        creg[0][r] = open_hx[(grow * 2 + 1) * HID + j0];
        creg[1][r] = open_hx[(grow * 2 + 1) * HID + j1];
    }
    // h0 into LDS as bf16
    for (int e = tid; e < ROWS_PER_BLK * HID; e += NTH) {
        int m = e >> 8, j = e & 255;
        hlds[m * HSTR + j] = f2bf16(open_hx[((row0 + m) * 2 + 0) * HID + j]);
    }
    // per-row constants
    if (tid < ROWS_PER_BLK) {
        int grow = row0 + tid;
        lsA[tid] = ls_probs[grow];
        opA[tid] = open_probs[grow];
        int os   = open_slices[grow];
        oblA[tid] = logf(p[os * 2 + 0]);
        oslA[tid] = logf(p[os * 2 + 1]);
    }
    float nd = 1.0f, lossm = 0.0f;   // sequential chain state (threads 0..31)
    __syncthreads();

    for (int ch = 0; ch < nch; ++ch) {
        // per-chunk precompute: normalized inputs + loss terms (2048 entries)
        #pragma unroll
        for (int i = 0; i < 4; ++i) {
            int e = tid * 4 + i;             // 0..2047
            int t = e >> 5, m = e & 31;
            int grow = row0 + m;
            int idx0 = inds[grow >> 4] + (grow & 15) + ch * CHUNK;
            float xref = input[2 * idx0];
            aalds[(t * ROWS_PER_BLK + m) * 2 + 0] = input[2 * (idx0 + t) + 0] - xref;
            aalds[(t * ROWS_PER_BLK + m) * 2 + 1] = input[2 * (idx0 + t) + 1] - xref;
            float p0 = p[2 * (idx0 + t) + 0];
            float p1 = p[2 * (idx0 + t) + 1];
            float term = lsA[m] * (logf(p1) - oblA[m]) + (1.0f - lsA[m]) * (oslA[m] - logf(p0));
            wterm[t * ROWS_PER_BLK + m] = opA[m] * term;
        }
        __syncthreads();

        for (int t = 0; t < CHUNK; ++t) {
            // Launder a 32-bit ZERO offset through volatile asm each timestep:
            // blocks CSE/LICM of weight loads across iterations while keeping
            // global address-space inference (global_load, not flat).
            unsigned int zoff = 0;
            asm volatile("" : "+s"(zoff));
            const unsigned short* wbp = whhb + zoff;

            // ---- 1) gates = h @ W_hh.T via WMMA bf16 (f32 accum) ----
            // Software pipeline, prefetch distance 2: three rotating half-k B
            // buffers; WMMA quartet g consumes loads issued at slot g-2.
            v8f acc[8] = {};
            const int arow = (16 * mt + lm) * HSTR;
            Frag A2[2];
            Frag Bb[3][4];

#define LOADA(dst, ks_) do {                                                   \
    int o1_ = (ks_) * 32 + 8 * lh;                                             \
    (dst).q[0] = *(const i4*)(&hlds[arow + o1_]);                              \
    (dst).q[1] = *(const i4*)(&hlds[arow + o1_ + 16]);                         \
} while (0)
#define LOADB(buf, g_) do {                                                    \
    _Pragma("unroll")                                                          \
    for (int ss = 0; ss < 4; ++ss) {                                           \
        const unsigned short* bp_ = wbp +                                      \
            ((wj + 8 * ((((g_) & 1) * 4) + ss)) * 16 + lm) * HID +             \
            (((g_) >> 1) * 32) + 8 * lh;                                       \
        (buf)[ss].q[0] = *(const i4*)(bp_);                                    \
        (buf)[ss].q[1] = *(const i4*)(bp_ + 16);                               \
    }                                                                          \
} while (0)
#define WMMA4(Aa, buf, h_) do {                                                \
    _Pragma("unroll")                                                          \
    for (int ss = 0; ss < 4; ++ss)                                             \
        acc[(h_) * 4 + ss] = __builtin_amdgcn_wmma_f32_16x16x32_bf16(          \
            false, (Aa).v, false, (buf)[ss].v, (short)0,                       \
            acc[(h_) * 4 + ss], false, false);                                 \
} while (0)

            LOADA(A2[0], 0);
            LOADB(Bb[0], 0);                       // prologue: groups 0 and 1
            LOADB(Bb[1], 1);
            #pragma unroll
            for (int g = 0; g < 16; ++g) {         // group g = (ks = g>>1, half = g&1)
                if (g + 2 < 16) LOADB(Bb[(g + 2) % 3], g + 2);
                const int ks = g >> 1;
                if ((g & 1) == 0 && ks < 7) LOADA(A2[(ks + 1) & 1], ks + 1);
                WMMA4(A2[ks & 1], Bb[g % 3], g & 1);
            }
#undef LOADA
#undef LOADB
#undef WMMA4

            // Pin the pipeline: [8 VMEM-read][8 VMEM-read] then 14x
            // [8 VMEM-read][4 WMMA] then [4 WMMA][4 WMMA] -> every WMMA
            // quartet trails its loads by two groups.
            __builtin_amdgcn_sched_group_barrier(0x020, 8, 0);
            __builtin_amdgcn_sched_group_barrier(0x020, 8, 0);
            #pragma unroll
            for (int g = 0; g < 14; ++g) {
                __builtin_amdgcn_sched_group_barrier(0x020, 8, 0);
                __builtin_amdgcn_sched_group_barrier(0x008, 4, 0);
            }
            __builtin_amdgcn_sched_group_barrier(0x008, 4, 0);
            __builtin_amdgcn_sched_group_barrier(0x008, 4, 0);

            // ---- 2) add input projection + bias, LSTM cell update (registers) ----
            f2 av[8];
            #pragma unroll
            for (int r = 0; r < 8; ++r)
                av[r] = *(const f2*)(&aalds[(t * ROWS_PER_BLK + 16 * mt + r + 8 * lh) * 2]);
            float hnew[2][8], part[8];
            #pragma unroll
            for (int r = 0; r < 8; ++r) {
                #pragma unroll
                for (int jj = 0; jj < 2; ++jj) {
                    float gi = acc[0 + jj][r] + av[r][0] * wih0[0 + jj] + av[r][1] * wih1[0 + jj] + bsum[0 + jj];
                    float gf = acc[2 + jj][r] + av[r][0] * wih0[2 + jj] + av[r][1] * wih1[2 + jj] + bsum[2 + jj];
                    float gg = acc[4 + jj][r] + av[r][0] * wih0[4 + jj] + av[r][1] * wih1[4 + jj] + bsum[4 + jj];
                    float go = acc[6 + jj][r] + av[r][0] * wih0[6 + jj] + av[r][1] * wih1[6 + jj] + bsum[6 + jj];
                    float c2 = sigm(gf) * creg[jj][r] + sigm(gi) * tanh_(gg);
                    creg[jj][r] = c2;
                    hnew[jj][r] = sigm(go) * tanh_(c2);
                }
                part[r] = hnew[0][r] * wo0 + hnew[1][r] * wo1;   // W_out partial dot
            }
            __syncthreads();   // all A-fragment reads of h(t) done

            // ---- 3) write h(t+1) to LDS (bf16) + butterfly-reduce output dot ----
            #pragma unroll
            for (int r = 0; r < 8; ++r) {
                int m = 16 * mt + r + 8 * lh;
                hlds[m * HSTR + j0] = f2bf16(hnew[0][r]);
                hlds[m * HSTR + j1] = f2bf16(hnew[1][r]);
                float pr = part[r];
                #pragma unroll
                for (int d = 1; d < 16; d <<= 1) pr += __shfl_xor(pr, d);
                if (lm == 0) scratch[wj * ROWS_PER_BLK + m] = pr;
            }
            __syncthreads();

            // ---- 4) sequential prob / cumprod / loss chain (one thread per row) ----
            if (tid < ROWS_PER_BLK) {
                float dot = b_out[0];
                #pragma unroll
                for (int w = 0; w < 8; ++w) dot += scratch[w * ROWS_PER_BLK + tid];
                float prob = sigm(dot);
                float wgt  = (t == 0) ? 1.0f : nd;   // reference weights t==0 by 1.0
                lossm += wterm[t * ROWS_PER_BLK + tid] * prob * wgt;
                nd *= (1.0f - prob);
            }
        }
        __syncthreads();   // protect aalds/wterm before next chunk rewrites them
    }

    // deterministic per-block loss partial (fixed-order butterfly, wave 0)
    if (tid < 32) {
        float v = lossm;
        #pragma unroll
        for (int d = 1; d < 32; d <<= 1) v += __shfl_xor(v, d);
        if (tid == 0) loss_part[blk] = v;
    }
}

// ---- deterministic final sum --------------------------------------------------
__global__ void finalize(const float* __restrict__ loss_part, float* __restrict__ out) {
    if (threadIdx.x == 0) {
        float s = 0.0f;
        for (int i = 0; i < NBLK; ++i) s += loss_part[i];
        out[0] = s;
    }
}

extern "C" void kernel_launch(void* const* d_in, const int* in_sizes, int n_in,
                              void* d_out, int out_size, void* d_ws, size_t ws_size,
                              hipStream_t stream) {
    (void)in_sizes; (void)n_in; (void)out_size; (void)ws_size;
    const int*   inds        = (const int*)  d_in[0];
    const float* input       = (const float*)d_in[1];
    const float* p           = (const float*)d_in[2];
    const float* ls_probs    = (const float*)d_in[3];
    const float* open_probs  = (const float*)d_in[4];
    const int*   open_slices = (const int*)  d_in[5];
    const float* open_hx     = (const float*)d_in[6];
    const float* W_ih        = (const float*)d_in[7];
    const float* W_hh        = (const float*)d_in[8];
    const float* b_ih        = (const float*)d_in[9];
    const float* b_hh        = (const float*)d_in[10];
    const float* W_out       = (const float*)d_in[11];
    const float* b_out       = (const float*)d_in[12];

    unsigned short* whhb   = (unsigned short*)d_ws;                       // 512 KB
    int*            nch    = (int*)((char*)d_ws + (size_t)NGATE * HID * 2);
    float*          lpart  = (float*)((char*)d_ws + (size_t)NGATE * HID * 2 + 64);

    prep_meta<<<1, 1024, 0, stream>>>(open_slices, nch);
    prep_whh<<<256, 256, 0, stream>>>(W_hh, whhb);
    lstm_main<<<NBLK, NTH, 0, stream>>>(inds, input, p, ls_probs, open_probs, open_slices,
                                        open_hx, W_ih, b_ih, b_hh, W_out, b_out,
                                        whhb, nch, lpart);
    finalize<<<1, 32, 0, stream>>>(lpart, (float*)d_out);
}
